// CrossST_fine_80367428043067
// MI455X (gfx1250) — matrix operand order repaired and checked by reading, not compiled, not run on hardware
//
#include <hip/hip_runtime.h>
#include <cmath>

typedef unsigned short u16;
typedef unsigned int   u32;
typedef unsigned long long u64;

typedef __attribute__((ext_vector_type(16))) __bf16 v16bf;
typedef __attribute__((ext_vector_type(8)))  float  v8f;

static constexpr int BB = 16;    // batch
static constexpr int TT = 12;    // time steps
static constexpr int NN = 2048;  // nodes
static constexpr int DD = 64;    // fine width
static constexpr int PDD = 256;  // pre width
static constexpr u64 PP = (u64)BB * NN;  // 32768 positions

#define PI_F 3.14159265358979323846f

__device__ __forceinline__ u16 f2bf(float f) {
  u32 u = __float_as_uint(f);
  u32 r = (u + 0x7FFFu + ((u >> 16) & 1u)) >> 16;  // RNE
  return (u16)r;
}
__device__ __forceinline__ float bf2f(u16 h) { return __uint_as_float(((u32)h) << 16); }

union FragBF { v16bf v; uint4 u[2]; };

// ---------------------------------------------------------------------------
// Generic bf16 WMMA GEMM:  D[col, m] = act( sum_k A[m,k] * B[col,k] + bias[m] ) + resid[col,m]
// A row-major [M,K] bf16 (lda), B row-major [Ncols,K] bf16 (ldb), D [Ncols,M] (ldd).
// blockIdx.z = batch with element strides sA/sB/sR/sD. act: 0 none, 1 relu.
// Wave computes 32(M) x 64(cols): 2 A-frags x 4 B-frags = 8 wmma per K-step,
// with a branch-free hot loop (prefetch region split off the K tail).
// ---------------------------------------------------------------------------
__global__ __launch_bounds__(128)
void gemm_bf16_kernel(const u16* __restrict__ A, int lda, u64 sA,
                      const u16* __restrict__ Bm, int ldb, u64 sB,
                      const float* __restrict__ bias,
                      const float* __restrict__ resid, int ldr, u64 sR,
                      float* __restrict__ outF, u16* __restrict__ outH, int ldd, u64 sD,
                      int M, int K, int Ncols, int act)
{
  const int lane = threadIdx.x & 31;
  const int wave = threadIdx.x >> 5;
  const int half = lane >> 4;
  const int r    = lane & 15;
  const u64 z    = blockIdx.z;
  const int m0   = blockIdx.y << 5;     // 32 output rows per block
  const int n0   = blockIdx.x * 256 + wave * 64;
  if (n0 >= Ncols) return;
  const u16* Ab  = A + z * sA;
  const u16* Bb2 = Bm + z * sB;

  v8f acc[2][4] = {};

  const int  mr0 = m0 + r;
  const int  mr1 = m0 + 16 + r;
  const bool ok0 = (mr0 < M);
  const bool ok1 = (mr1 < M);
  const u16* arow0 = Ab + (u64)(ok0 ? mr0 : 0) * (u64)lda;
  const u16* arow1 = Ab + (u64)(ok1 ? mr1 : 0) * (u64)lda;

#define GEMM_BODY(PF, GUARD)                                                     \
  {                                                                              \
    FragBF a0, a1;                                                               \
    { const uint4* pa = (const uint4*)(arow0 + k0);                              \
      a0.u[0] = pa[half]; a0.u[1] = pa[2 + half]; }                              \
    { const uint4* pa = (const uint4*)(arow1 + k0);                              \
      a1.u[0] = pa[half]; a1.u[1] = pa[2 + half]; }                              \
    if (GUARD) {                                                                 \
      if (!ok0) { a0.u[0] = make_uint4(0u,0u,0u,0u); a0.u[1] = make_uint4(0u,0u,0u,0u); } \
      if (!ok1) { a1.u[0] = make_uint4(0u,0u,0u,0u); a1.u[1] = make_uint4(0u,0u,0u,0u); } \
    }                                                                            \
    _Pragma("unroll")                                                            \
    for (int j = 0; j < 4; ++j) {                                                \
      const u64 col = (u64)(n0 + (j << 4) + r);                                  \
      const u16* brow = Bb2 + col * (u64)ldb + k0;                               \
      if (PF) __builtin_prefetch((const void*)(brow + 128), 0, 3);               \
      const uint4* pb = (const uint4*)(brow);                                    \
      FragBF bfrag;                                                              \
      bfrag.u[0] = pb[half];                                                     \
      bfrag.u[1] = pb[2 + half];                                                 \
      acc[0][j] = __builtin_amdgcn_wmma_f32_16x16x32_bf16(                       \
          false, a0.v, false, bfrag.v, (short)0, acc[0][j], false, false);       \
      acc[1][j] = __builtin_amdgcn_wmma_f32_16x16x32_bf16(                       \
          false, a1.v, false, bfrag.v, (short)0, acc[1][j], false, false);       \
    }                                                                            \
  }

  if (m0 + 32 <= M) {
    // fast path: no per-lane M guards; prefetch region split off the K tail
    const int kPF = K - 128;
    int k0 = 0;
    for (; k0 < kPF; k0 += 32) GEMM_BODY(true, false)
    for (; k0 < K; k0 += 32)  GEMM_BODY(false, false)
  } else {
    for (int k0 = 0; k0 < K; k0 += 32) GEMM_BODY(false, true)
  }
#undef GEMM_BODY

#pragma unroll
  for (int mi = 0; mi < 2; ++mi) {
    const int mbase = m0 + (mi << 4) + (half << 3);
#pragma unroll
    for (int j = 0; j < 4; ++j) {
      const u64 col = (u64)(n0 + (j << 4) + r);
      float vals[8];
#pragma unroll
      for (int vi = 0; vi < 8; ++vi) {
        int m = mbase + vi;
        float xv = acc[mi][j][vi];
        if (bias && m < M)  xv += bias[m];
        if (act == 1)       xv = fmaxf(xv, 0.f);
        if (resid && m < M) xv += resid[z * sR + col * (u64)ldr + m];
        vals[vi] = xv;
      }
      if (outF) {
        float* po = outF + z * sD + col * (u64)ldd + mbase;
        if (mbase + 8 <= M) {
          *(float4*)(po)     = make_float4(vals[0], vals[1], vals[2], vals[3]);
          *(float4*)(po + 4) = make_float4(vals[4], vals[5], vals[6], vals[7]);
        } else {
          for (int vi = 0; vi < 8; ++vi) if (mbase + vi < M) po[vi] = vals[vi];
        }
      }
      if (outH) {
        u16* ph = outH + z * sD + col * (u64)ldd + mbase;
        if (mbase + 8 <= M) {
          uint4 up;
          up.x = (u32)f2bf(vals[0]) | ((u32)f2bf(vals[1]) << 16);
          up.y = (u32)f2bf(vals[2]) | ((u32)f2bf(vals[3]) << 16);
          up.z = (u32)f2bf(vals[4]) | ((u32)f2bf(vals[5]) << 16);
          up.w = (u32)f2bf(vals[6]) | ((u32)f2bf(vals[7]) << 16);
          *(uint4*)ph = up;
        } else {
          for (int vi = 0; vi < 8; ++vi) if (mbase + vi < M) ph[vi] = f2bf(vals[vi]);
        }
      }
    }
  }
}

// --------------------------- small utility kernels --------------------------

__global__ void cvt_kernel(const float* __restrict__ X, u16* __restrict__ Y, u64 n) {
  u64 i = (u64)blockIdx.x * blockDim.x + threadIdx.x;
  if (i < n) Y[i] = f2bf(X[i]);
}

// W [d,d,1,5] -> Wf[o, k*d + c]  (flattened-K im2col order), bf16
__global__ void tcnw_kernel(const float* __restrict__ W, u16* __restrict__ Wf, int d, u64 total) {
  u64 i = (u64)blockIdx.x * blockDim.x + threadIdx.x;
  if (i >= total) return;
  int per = d * 5;
  u64 o = i / per;
  u32 rem = (u32)(i - o * per);
  int k = rem / d, c = rem - k * d;
  Wf[o * (u64)per + (u64)k * d + c] = f2bf(W[(o * (u64)d + c) * 5 + k]);
}

// Combined patch-embed + (rfft * cw -> irfft) operator:
//   G[(tt*d + c), (t*3 + i)] = K_c[tt,t] * eW[i,c]   (bf16, K padded 36->64 with zeros)
//   Kbias[tt*d + c]          = sum_t K_c[tt,t] * tem[t,c]
__global__ void gop_kernel(const float* __restrict__ cw, const float* __restrict__ eW,
                           const float* __restrict__ tem,
                           u16* __restrict__ G, float* __restrict__ Kbias, int d)
{
  int idx = blockIdx.x * blockDim.x + threadIdx.x;  // over d*12
  if (idx >= d * 12) return;
  int tt = idx / d;
  int c  = idx - tt * d;
  float kc[12];
#pragma unroll
  for (int t = 0; t < 12; ++t) {
    int delta = tt - t;
    float s = cw[(c * 7 + 0) * 2 + 0];  // f=0: real part only
    for (int f = 1; f <= 5; ++f) {
      float a = cw[(c * 7 + f) * 2 + 0];
      float b = cw[(c * 7 + f) * 2 + 1];
      float th = 2.f * PI_F * (float)(f * delta) / 12.f;
      s += 2.f * (a * __cosf(th) - b * __sinf(th));
    }
    s += cw[(c * 7 + 6) * 2 + 0] * ((delta & 1) ? -1.f : 1.f);  // f=T/2
    kc[t] = s / 12.f;
  }
  u16* row = G + (u64)idx * 64;
#pragma unroll
  for (int t = 0; t < 12; ++t) {
    float w0 = eW[0 * d + c], w1 = eW[1 * d + c], w2 = eW[2 * d + c];
    row[t * 3 + 0] = f2bf(kc[t] * w0);
    row[t * 3 + 1] = f2bf(kc[t] * w1);
    row[t * 3 + 2] = f2bf(kc[t] * w2);
  }
  for (int k = 36; k < 64; ++k) row[k] = 0;
  float bs = 0.f;
#pragma unroll
  for (int t = 0; t < 12; ++t) bs += kc[t] * tem[t * d + c];
  Kbias[idx] = bs;
}

// x [B,T,N,3] -> Xb [P, 64] bf16 (K-padded input patches, shared by both branches)
__global__ void xbuild_kernel(const float* __restrict__ x, u16* __restrict__ Xb, u64 total) {
  u64 i = (u64)blockIdx.x * blockDim.x + threadIdx.x;
  if (i >= total) return;
  u64 p = i >> 6;
  int k = (int)(i & 63);
  if (k < 36) {
    int t = k / 3, ii = k - t * 3;
    int b = (int)(p / NN), n = (int)(p - (u64)b * NN);
    Xb[i] = f2bf(x[(((u64)b * TT + t) * NN + n) * 3 + ii]);
  } else {
    Xb[i] = 0;
  }
}

// zero the left-pad slot of xp and extract the TCN residual (= h1 at t=T-1)
__global__ void padres_kernel(u16* __restrict__ xp, float* __restrict__ resid,
                              int d, u64 total /* P*d */) {
  u64 i = (u64)blockIdx.x * blockDim.x + threadIdx.x;
  if (i >= total) return;
  u64 p = i / d;
  int c = (int)(i - p * d);
  xp[p * (u64)(13 * d) + c] = 0;
  resid[i] = bf2f(xp[(p * 13u + 12) * (u64)d + c]);
}

// Row softmax (scale applied pre-softmax), bf16 out, blockDim == d (pow2 <= 256)
__global__ void softmax_rows_kernel(const float* __restrict__ X, u16* __restrict__ Y,
                                    int d, float scale)
{
  u64 row = blockIdx.x;
  int c = threadIdx.x;
  __shared__ float red[256];
  float v = X[row * (u64)d + c] * scale;
  red[c] = v; __syncthreads();
  for (int o = blockDim.x >> 1; o > 0; o >>= 1) {
    if (c < o) red[c] = fmaxf(red[c], red[c + o]);
    __syncthreads();
  }
  float mx = red[0]; __syncthreads();
  float e = __expf(v - mx);
  red[c] = e; __syncthreads();
  for (int o = blockDim.x >> 1; o > 0; o >>= 1) {
    if (c < o) red[c] += red[c + o];
    __syncthreads();
  }
  Y[row * (u64)d + c] = f2bf(e / red[0]);
}

// softmax over d of bank[n,d], stored transposed keyT[c*N + n] (bf16)
__global__ void softmax_keyT_kernel(const float* __restrict__ bank, u16* __restrict__ keyT,
                                    int d, float scale)
{
  u64 n = blockIdx.x;
  int c = threadIdx.x;
  __shared__ float red[256];
  float v = bank[n * (u64)d + c] * scale;
  red[c] = v; __syncthreads();
  for (int o = blockDim.x >> 1; o > 0; o >>= 1) {
    if (c < o) red[c] = fmaxf(red[c], red[c + o]);
    __syncthreads();
  }
  float mx = red[0]; __syncthreads();
  float e = __expf(v - mx);
  red[c] = e; __syncthreads();
  for (int o = blockDim.x >> 1; o > 0; o >>= 1) {
    if (c < o) red[c] += red[c + o];
    __syncthreads();
  }
  keyT[(u64)c * NN + n] = f2bf(e / red[0]);
}

// v [P,d] bf16 -> vT [(b*d + y)*N + n] bf16
__global__ void vtrans_kernel(const u16* __restrict__ V, u16* __restrict__ VT, int d, u64 total) {
  u64 i = (u64)blockIdx.x * blockDim.x + threadIdx.x;
  if (i >= total) return;
  u64 p = i / d;
  int y = (int)(i - p * d);
  int b = (int)(p / NN), n = (int)(p - (u64)b * NN);
  VT[((u64)b * d + y) * NN + n] = V[i];
}

// partial sums for layernorm over (C,N) per batch; X [P,d] with batch-contiguous rows
__global__ void ln_partial_kernel(const float* __restrict__ X, float2* __restrict__ part,
                                  u64 perBatch, int S)
{
  int b = blockIdx.y, s = blockIdx.x;
  u64 base = (u64)b * perBatch;
  u64 chunk = (perBatch + S - 1) / S;
  u64 beg = (u64)s * chunk;
  u64 end = beg + chunk; if (end > perBatch) end = perBatch;
  float sm = 0.f, sq = 0.f;
  for (u64 i = beg + threadIdx.x; i < end; i += blockDim.x) {
    float v = X[base + i]; sm += v; sq += v * v;
  }
  __shared__ float rs[256], rq[256];
  int t = threadIdx.x;
  rs[t] = sm; rq[t] = sq; __syncthreads();
  for (int o = 128; o > 0; o >>= 1) {
    if (t < o) { rs[t] += rs[t + o]; rq[t] += rq[t + o]; }
    __syncthreads();
  }
  if (t == 0) part[(u64)b * S + s] = make_float2(rs[0], rq[0]);
}

__global__ void ln_finalize_kernel(const float2* __restrict__ part, float2* __restrict__ mv,
                                   int S, float invCount)
{
  int b = blockIdx.x, t = threadIdx.x;
  __shared__ float rs[256], rq[256];
  float sm = 0.f, sq = 0.f;
  for (int i = t; i < S; i += blockDim.x) {
    float2 p = part[(u64)b * S + i]; sm += p.x; sq += p.y;
  }
  rs[t] = sm; rq[t] = sq; __syncthreads();
  for (int o = 128; o > 0; o >>= 1) {
    if (t < o) { rs[t] += rs[t + o]; rq[t] += rq[t + o]; }
    __syncthreads();
  }
  if (t == 0) {
    float mean = rs[0] * invCount;
    float var = rq[0] * invCount - mean * mean;
    mv[b] = make_float2(mean, rsqrtf(var + 1e-5f));
  }
}

__global__ void ln_apply_kernel(const float* __restrict__ X, const float2* __restrict__ mv,
                                const float* __restrict__ gamma, const float* __restrict__ beta,
                                float* __restrict__ outF, u16* __restrict__ outH,
                                int d, u64 total)
{
  u64 i = (u64)blockIdx.x * blockDim.x + threadIdx.x;
  if (i >= total) return;
  u64 p = i / d;
  int c = (int)(i - p * d);
  int b = (int)(p / NN), n = (int)(p - (u64)b * NN);
  float2 m = mv[b];
  float y = (X[i] - m.x) * m.y * gamma[(u64)c * NN + n] + beta[(u64)c * NN + n];
  if (outF) outF[i] = y;
  if (outH) outH[i] = f2bf(y);
}

__global__ void glu_kernel(const float* __restrict__ A, const float* __restrict__ Bq,
                           u16* __restrict__ Y, u64 total)
{
  u64 i = (u64)blockIdx.x * blockDim.x + threadIdx.x;
  if (i >= total) return;
  float sg = 1.f / (1.f + __expf(-Bq[i]));
  Y[i] = f2bf(A[i] * sg);
}

// st = concat( ht2 + filt_t*ph_t + bias_t , hs2 + filt_s*ph_s + bias_s ) over 2*PD channels
__global__ void fuse_st_kernel(const float* __restrict__ ht2, const float* __restrict__ hs2,
                               const float* __restrict__ ph_t, const float* __restrict__ ph_s,
                               const float* __restrict__ filt_t, const float* __restrict__ bias_t,
                               const float* __restrict__ filt_s, const float* __restrict__ bias_s,
                               float* __restrict__ stF, u16* __restrict__ stH, u64 total)
{
  u64 i = (u64)blockIdx.x * blockDim.x + threadIdx.x;
  if (i >= total) return;
  u64 p = i / (2 * PDD);
  int c = (int)(i - p * (2 * PDD));
  int n = (int)(p % NN);
  float v;
  if (c < PDD)
    v = ht2[p * PDD + c] + filt_t[(u64)c * NN + n] * ph_t[p * PDD + c] + bias_t[(u64)c * NN + n];
  else {
    int cc = c - PDD;
    v = hs2[p * PDD + cc] + filt_s[(u64)cc * NN + n] * ph_s[p * PDD + cc] + bias_s[(u64)cc * NN + n];
  }
  stF[i] = v;
  stH[i] = f2bf(v);
}

// ytmp [P,12] -> out [B,T,N]
__global__ void outperm_kernel(const float* __restrict__ Yt, float* __restrict__ out, u64 total) {
  u64 i = (u64)blockIdx.x * blockDim.x + threadIdx.x;
  if (i >= total) return;
  u64 bt = i / NN;
  int n = (int)(i - bt * NN);
  int t = (int)(bt % TT);
  u64 b = bt / TT;
  out[i] = Yt[(b * NN + n) * TT + t];
}

// ------------------------------- host side ---------------------------------

static inline unsigned cdivu(u64 a, u64 b) { return (unsigned)((a + b - 1) / b); }

static void gemm(hipStream_t st, const u16* A, int lda, u64 sA,
                 const u16* Bm, int ldb, u64 sB,
                 const float* bias, const float* resid, int ldr, u64 sR,
                 float* outF, u16* outH, int ldd, u64 sD,
                 int M, int K, int Ncols, int act, int batches)
{
  dim3 g(cdivu(Ncols, 256), cdivu(M, 32), batches);
  gemm_bf16_kernel<<<g, 128, 0, st>>>(A, lda, sA, Bm, ldb, sB, bias, resid, ldr, sR,
                                      outF, outH, ldd, sD, M, K, Ncols, act);
}

struct BranchParams {
  const float *embed_w, *tem, *cw;
  const float *tw[3], *tb[3];
  const float *s1, *s1b, *s2, *s2b;
  const float *q_w, *q_b, *v_w, *v_b, *cat_w, *cat_b, *bank;
  const float *g1_w, *g1_b, *g2_w, *g2_b, *g3_w, *g3_b, *fc_w, *fc_b;
};

static void run_branch(hipStream_t st, const u16* Xb, const BranchParams& bp, int d,
                       char* arena, float* h_t, u16* h_t_h, float* h_s, u16* h_s_h)
{
  char* cur = arena;
  auto alloc = [&](u64 bytes) -> void* {
    void* p = (void*)cur;
    cur += (bytes + 255) & ~(u64)255;
    return p;
  };
  const u64 Pd = PP * (u64)d;
  const float scale = 1.0f / sqrtf((float)d);

  u16* wf[3];
  for (int k = 0; k < 3; ++k) wf[k] = (u16*)alloc((u64)d * 5 * d * 2);
  u16* wq   = (u16*)alloc((u64)d * d * 2);
  u16* wv   = (u16*)alloc((u64)d * d * 2);
  u16* wcat = (u16*)alloc((u64)d * d * 2);
  u16* wg1  = (u16*)alloc((u64)d * d * 2);
  u16* wg2  = (u16*)alloc((u64)d * d * 2);
  u16* wg3  = (u16*)alloc((u64)d * d * 2);
  u16* wfc  = (u16*)alloc((u64)d * d * 2);
  u16* G      = (u16*)alloc((u64)d * 12 * 64 * 2);
  float* Kbias = (float*)alloc((u64)d * 12 * 4);
  u16* xp     = (u16*)alloc(PP * 13 * (u64)d * 2);
  float* rbuf = (float*)alloc(Pd * 4);
  u16* y1     = (u16*)alloc(PP * 9 * (u64)d * 2);
  u16* y2     = (u16*)alloc(PP * 5 * (u64)d * 2);
  float* qbuf = (float*)alloc(Pd * 4);
  u16* qsm    = (u16*)alloc(Pd * 2);
  u16* vh     = (u16*)alloc(Pd * 2);
  u16* vT     = (u16*)alloc(Pd * 2);
  u16* keyT   = (u16*)alloc((u64)d * NN * 2);
  u16* kvT    = (u16*)alloc((u64)BB * d * d * 2);
  u16* obuf   = (u16*)alloc(Pd * 2);
  float* hsb  = (float*)alloc(Pd * 4);
  float2* part = (float2*)alloc((u64)BB * 256 * 8);
  float2* mv   = (float2*)alloc((u64)BB * 8);
  float* hsn  = (float*)alloc(Pd * 4);
  u16* hsnh   = (u16*)alloc(Pd * 2);
  float* g1f  = (float*)alloc(Pd * 4);
  float* g2f  = (float*)alloc(Pd * 4);
  u16* gh     = (u16*)alloc(Pd * 2);
  float* hf   = (float*)alloc(Pd * 4);
  float* hf2  = (float*)alloc(Pd * 4);

  const u64 wdd = (u64)d * d;
  for (int k = 0; k < 3; ++k)
    tcnw_kernel<<<cdivu(wdd * 5, 256), 256, 0, st>>>(bp.tw[k], wf[k], d, wdd * 5);
  cvt_kernel<<<cdivu(wdd, 256), 256, 0, st>>>(bp.q_w, wq, wdd);
  cvt_kernel<<<cdivu(wdd, 256), 256, 0, st>>>(bp.v_w, wv, wdd);
  cvt_kernel<<<cdivu(wdd, 256), 256, 0, st>>>(bp.cat_w, wcat, wdd);
  cvt_kernel<<<cdivu(wdd, 256), 256, 0, st>>>(bp.g1_w, wg1, wdd);
  cvt_kernel<<<cdivu(wdd, 256), 256, 0, st>>>(bp.g2_w, wg2, wdd);
  cvt_kernel<<<cdivu(wdd, 256), 256, 0, st>>>(bp.g3_w, wg3, wdd);
  cvt_kernel<<<cdivu(wdd, 256), 256, 0, st>>>(bp.fc_w, wfc, wdd);
  gop_kernel<<<cdivu((u64)d * 12, 256), 256, 0, st>>>(bp.cw, bp.embed_w, bp.tem, G, Kbias, d);

  // Patch-embed + frequency filter as one WMMA GEMM: writes xp slots 1..12
  gemm(st, G, 64, 0, Xb, 64, 0, Kbias, nullptr, 0, 0,
       nullptr, xp + d, 13 * d, 0, 12 * d, 64, (int)PP, 0, 1);
  padres_kernel<<<cdivu(Pd, 256), 256, 0, st>>>(xp, rbuf, d, Pd);

  // TCN: implicit im2col; blockIdx.z = output time step (strides in elements of d)
  gemm(st, wf[0], 5 * d, 0, xp, 13 * d, (u64)d, bp.tb[0], nullptr, 0, 0,
       nullptr, y1, 9 * d, (u64)d, d, 5 * d, (int)PP, 1, 9);
  gemm(st, wf[1], 5 * d, 0, y1, 9 * d, (u64)d, bp.tb[1], nullptr, 0, 0,
       nullptr, y2, 5 * d, (u64)d, d, 5 * d, (int)PP, 1, 5);
  gemm(st, wf[2], 5 * d, 0, y2, 5 * d, 0, bp.tb[2], rbuf, d, 0,
       h_t, h_t_h, d, 0, d, 5 * d, (int)PP, 1, 1);

  // Attention (linear attention with softmax'd key bank)
  gemm(st, wq, d, 0, h_t_h, d, 0, bp.q_b, nullptr, 0, 0, qbuf, nullptr, d, 0, d, d, (int)PP, 1, 1);
  gemm(st, wv, d, 0, h_t_h, d, 0, bp.v_b, nullptr, 0, 0, nullptr, vh, d, 0, d, d, (int)PP, 1, 1);
  softmax_rows_kernel<<<(unsigned)PP, d, 0, st>>>(qbuf, qsm, d, scale);
  softmax_keyT_kernel<<<(unsigned)NN, d, 0, st>>>(bp.bank, keyT, d, scale);
  vtrans_kernel<<<cdivu(Pd, 256), 256, 0, st>>>(vh, vT, d, Pd);
  // kv[b,x,y] = sum_n key[n,x] v[b,n,y]   (stored kvT[b][y][x])
  gemm(st, keyT, NN, 0, vT, NN, 0, nullptr, nullptr, 0, 0,
       nullptr, kvT, d, 0, d, NN, BB * d, 0, 1);
  // o[b,n,y] = sum_x q[b,n,x] kv[b,x,y]  (batched over z)
  gemm(st, kvT, d, (u64)d * d, qsm, d, (u64)NN * d, nullptr, nullptr, 0, 0,
       nullptr, obuf, d, (u64)NN * d, d, d, NN, 0, BB);
  // hs = h + relu(Wcat o + b)
  gemm(st, wcat, d, 0, obuf, d, 0, bp.cat_b, h_t, d, 0, hsb, nullptr, d, 0, d, d, (int)PP, 1, 1);

  // LayerNorm over (C,N) per batch, scale s1
  const u64 perBatch = (u64)NN * d;
  ln_partial_kernel<<<dim3(256, BB), 256, 0, st>>>(hsb, part, perBatch, 256);
  ln_finalize_kernel<<<BB, 256, 0, st>>>(part, mv, 256, 1.f / (float)perBatch);
  ln_apply_kernel<<<cdivu(Pd, 256), 256, 0, st>>>(hsb, mv, bp.s1, bp.s1b, hsn, hsnh, d, Pd);

  // Gate: g = (W1 hsn + b1) * sigmoid(W2 hsn + b2);  hf = hsn + W3 g + b3
  gemm(st, wg1, d, 0, hsnh, d, 0, bp.g1_b, nullptr, 0, 0, g1f, nullptr, d, 0, d, d, (int)PP, 0, 1);
  gemm(st, wg2, d, 0, hsnh, d, 0, bp.g2_b, nullptr, 0, 0, g2f, nullptr, d, 0, d, d, (int)PP, 0, 1);
  glu_kernel<<<cdivu(Pd, 256), 256, 0, st>>>(g1f, g2f, gh, Pd);
  gemm(st, wg3, d, 0, gh, d, 0, bp.g3_b, hsn, d, 0, hf, nullptr, d, 0, d, d, (int)PP, 0, 1);

  // LayerNorm 2 (scale s2), then h_s = hf2 + Wfc h + bfc
  ln_partial_kernel<<<dim3(256, BB), 256, 0, st>>>(hf, part, perBatch, 256);
  ln_finalize_kernel<<<BB, 256, 0, st>>>(part, mv, 256, 1.f / (float)perBatch);
  ln_apply_kernel<<<cdivu(Pd, 256), 256, 0, st>>>(hf, mv, bp.s2, bp.s2b, hf2, nullptr, d, Pd);
  gemm(st, wfc, d, 0, h_t_h, d, 0, bp.fc_b, hf2, d, 0, h_s, h_s_h, d, 0, d, d, (int)PP, 0, 1);
}

extern "C" void kernel_launch(void* const* d_in, const int* in_sizes, int n_in,
                              void* d_out, int out_size, void* d_ws, size_t ws_size,
                              hipStream_t stream)
{
  (void)in_sizes; (void)n_in; (void)out_size; (void)ws_size;
  int i = 0;
  const float* x = (const float*)d_in[i++];
  auto readBranch = [&](BranchParams& bp) {
    bp.embed_w = (const float*)d_in[i++];
    bp.tem     = (const float*)d_in[i++];
    bp.cw      = (const float*)d_in[i++];
    for (int k = 0; k < 3; ++k) {
      bp.tw[k] = (const float*)d_in[i++];
      bp.tb[k] = (const float*)d_in[i++];
    }
    bp.s1 = (const float*)d_in[i++]; bp.s1b = (const float*)d_in[i++];
    bp.s2 = (const float*)d_in[i++]; bp.s2b = (const float*)d_in[i++];
    bp.q_w = (const float*)d_in[i++]; bp.q_b = (const float*)d_in[i++];
    bp.v_w = (const float*)d_in[i++]; bp.v_b = (const float*)d_in[i++];
    bp.cat_w = (const float*)d_in[i++]; bp.cat_b = (const float*)d_in[i++];
    bp.bank = (const float*)d_in[i++];
    bp.g1_w = (const float*)d_in[i++]; bp.g1_b = (const float*)d_in[i++];
    bp.g2_w = (const float*)d_in[i++]; bp.g2_b = (const float*)d_in[i++];
    bp.g3_w = (const float*)d_in[i++]; bp.g3_b = (const float*)d_in[i++];
    bp.fc_w = (const float*)d_in[i++]; bp.fc_b = (const float*)d_in[i++];
  };
  BranchParams pre, fine;
  readBranch(pre);
  readBranch(fine);
  const float* fct_w = (const float*)d_in[i++]; const float* fct_b = (const float*)d_in[i++];
  const float* fcs_w = (const float*)d_in[i++]; const float* fcs_b = (const float*)d_in[i++];
  const float* filter_t = (const float*)d_in[i++]; const float* bias_t = (const float*)d_in[i++];
  const float* filter_s = (const float*)d_in[i++]; const float* bias_s = (const float*)d_in[i++];
  const float* m1_w = (const float*)d_in[i++]; const float* m1_b = (const float*)d_in[i++];
  const float* m2_w = (const float*)d_in[i++]; const float* m2_b = (const float*)d_in[i++];
  const float* p_w = (const float*)d_in[i++]; const float* p_b = (const float*)d_in[i++];

  // Persistent region of workspace (survives across phases), then reusable arena.
  char* cur0 = (char*)d_ws;
  auto alloc0 = [&](u64 bytes) -> void* {
    void* p = (void*)cur0;
    cur0 += (bytes + 255) & ~(u64)255;
    return p;
  };
  float* ph_t  = (float*)alloc0(PP * PDD * 4);
  float* ph_s  = (float*)alloc0(PP * PDD * 4);
  u16*   ph_th = (u16*)alloc0(PP * PDD * 2);
  float* htf   = (float*)alloc0(PP * DD * 4);
  u16*   htf_h = (u16*)alloc0(PP * DD * 2);
  float* hsf   = (float*)alloc0(PP * DD * 4);
  u16*   hsf_h = (u16*)alloc0(PP * DD * 2);
  u16*   Xb    = (u16*)alloc0(PP * 64 * 2);
  char* arena = cur0;

  // Shared K-padded input patches for the embed GEMM of both branches
  xbuild_kernel<<<cdivu(PP * 64, 256), 256, 0, stream>>>(x, Xb, PP * 64);

  // Two branches (pre branch: no_grad is forward-identical)
  run_branch(stream, Xb, pre, PDD, arena, ph_t, ph_th, ph_s, nullptr);
  run_branch(stream, Xb, fine, DD, arena, htf, htf_h, hsf, hsf_h);

  // Fusion head (arena is reused)
  char* cur = arena;
  auto alloc = [&](u64 bytes) -> void* {
    void* p = (void*)cur;
    cur += (bytes + 255) & ~(u64)255;
    return p;
  };
  u16* wfct = (u16*)alloc((u64)PDD * DD * 2);
  u16* wfcs = (u16*)alloc((u64)PDD * DD * 2);
  u16* wm1  = (u16*)alloc((u64)512 * 512 * 2);
  u16* wm2  = (u16*)alloc((u64)512 * 512 * 2);
  u16* wpw  = (u16*)alloc((u64)TT * 512 * 2);
  float* ht2 = (float*)alloc(PP * PDD * 4);
  float* hs2 = (float*)alloc(PP * PDD * 4);
  float* stF = (float*)alloc(PP * 512 * 4);
  u16*   stH = (u16*)alloc(PP * 512 * 2);
  u16*   m1h = (u16*)alloc(PP * 512 * 2);
  u16*   mH  = (u16*)alloc(PP * 512 * 2);
  float* ytmp = (float*)alloc(PP * TT * 4);

  cvt_kernel<<<cdivu((u64)PDD * DD, 256), 256, 0, stream>>>(fct_w, wfct, (u64)PDD * DD);
  cvt_kernel<<<cdivu((u64)PDD * DD, 256), 256, 0, stream>>>(fcs_w, wfcs, (u64)PDD * DD);
  cvt_kernel<<<cdivu((u64)512 * 512, 256), 256, 0, stream>>>(m1_w, wm1, (u64)512 * 512);
  cvt_kernel<<<cdivu((u64)512 * 512, 256), 256, 0, stream>>>(m2_w, wm2, (u64)512 * 512);
  cvt_kernel<<<cdivu((u64)TT * 512, 256), 256, 0, stream>>>(p_w, wpw, (u64)TT * 512);

  gemm(stream, wfct, DD, 0, htf_h, DD, 0, fct_b, nullptr, 0, 0,
       ht2, nullptr, PDD, 0, PDD, DD, (int)PP, 1, 1);
  gemm(stream, wfcs, DD, 0, hsf_h, DD, 0, fcs_b, nullptr, 0, 0,
       hs2, nullptr, PDD, 0, PDD, DD, (int)PP, 1, 1);
  fuse_st_kernel<<<cdivu(PP * 512, 256), 256, 0, stream>>>(ht2, hs2, ph_t, ph_s,
      filter_t, bias_t, filter_s, bias_s, stF, stH, PP * 512);
  gemm(stream, wm1, 512, 0, stH, 512, 0, m1_b, nullptr, 0, 0,
       nullptr, m1h, 512, 0, 512, 512, (int)PP, 1, 1);
  gemm(stream, wm2, 512, 0, m1h, 512, 0, m2_b, stF, 512, 0,
       nullptr, mH, 512, 0, 512, 512, (int)PP, 0, 1);
  gemm(stream, wpw, 512, 0, mH, 512, 0, p_b, nullptr, 0, 0,
       ytmp, nullptr, TT, 0, TT, 512, (int)PP, 0, 1);
  outperm_kernel<<<cdivu((u64)BB * TT * NN, 256), 256, 0, stream>>>(ytmp, (float*)d_out,
      (u64)BB * TT * NN);
}
